// MovementUnitSelHead_17549236371833
// MI455X (gfx1250) — matrix operand order repaired and checked by reading, not compiled, not run on hardware
//
#include <hip/hip_runtime.h>

typedef __bf16 bf16_t;
typedef __attribute__((ext_vector_type(16))) __bf16 v16bf;
typedef __attribute__((ext_vector_type(8)))  __bf16 v8bf;
typedef __attribute__((ext_vector_type(8)))  float  v8f;
typedef int v4i_ __attribute__((vector_size(16)));

#define D_    1024
#define U_    4096
#define H_    16
#define HD_   64
#define L_    4
#define HID_  512
#define KPADI 1056   // D+1 padded to multiple of 32

// ---------------------------------------------------------------- CDNA5 helpers

__device__ __forceinline__ v8f wmma_bf16(v16bf a, v16bf b, v8f c) {
  // D = A(16x32 bf16) * B(32x16 bf16) + C(16x16 f32)
  return __builtin_amdgcn_wmma_f32_16x16x32_bf16(
      false, a, false, b, (short)0, c, false, false);
}

// Async global->LDS 16B copy (ASYNCcnt tracked). Per-lane addresses.
__device__ __forceinline__ void async_load_b128(void* lds, const void* gsrc) {
#if __has_builtin(__builtin_amdgcn_global_load_async_to_lds_b128)
  typedef __attribute__((address_space(1))) v4i_* gp_t;
  typedef __attribute__((address_space(3))) v4i_* lp_t;
  gp_t g = (gp_t)(unsigned long long)gsrc;
  lp_t l = (lp_t)(unsigned int)(unsigned long long)lds;
  __builtin_amdgcn_global_load_async_to_lds_b128(g, l, 0, 0);
#else
  unsigned int       l32 = (unsigned int)(unsigned long long)lds;
  unsigned long long g64 = (unsigned long long)gsrc;
  asm volatile("global_load_async_to_lds_b128 %0, %1, off"
               :: "v"(l32), "v"(g64) : "memory");
#endif
}

__device__ __forceinline__ void wait_async0() {
#if __has_builtin(__builtin_amdgcn_s_wait_asynccnt)
  __builtin_amdgcn_s_wait_asynccnt(0);
#else
  asm volatile("s_wait_asynccnt 0x0" ::: "memory");
#endif
}

// A-operand fragment (16x32 bf16, row major, lane=M, K split 0-7 / 8-15 per half)
__device__ __forceinline__ v16bf load_fragA(const bf16_t* base, int ld) {
  const int lane = threadIdx.x & 31;
  const int m    = lane & 15;
  const int koff = (lane >> 4) << 3;            // 0 or 8
  union { v16bf v; v8bf h[2]; } f;
  f.h[0] = *reinterpret_cast<const v8bf*>(base + m * ld + koff);
  f.h[1] = *reinterpret_cast<const v8bf*>(base + m * ld + 16 + koff);
  return f.v;
}

// B-operand fragment (32x16 bf16 as Bt[n][k] row major, lane=N, K split 0-7 / 8-15 per half)
__device__ __forceinline__ v16bf load_fragB(const bf16_t* base, int ld) {
  const int lane = threadIdx.x & 31;
  const int n  = lane & 15;
  const int kh = (lane >> 4) << 3;              // 0 or 8
  union { v16bf v; v8bf h[2]; } f;
  f.h[0] = *reinterpret_cast<const v8bf*>(base + n * ld + kh);
  f.h[1] = *reinterpret_cast<const v8bf*>(base + n * ld + 16 + kh);
  return f.v;
}

// ---------------------------------------------------------------- weight convert (pad K)

__global__ __launch_bounds__(256) void convw_kernel(const float* __restrict__ W,
                                                    bf16_t* __restrict__ Wb,
                                                    int N, int K, int Kpad) {
  size_t total = (size_t)N * Kpad;
  for (size_t i = (size_t)blockIdx.x * 256 + threadIdx.x; i < total;
       i += (size_t)gridDim.x * 256) {
    int n = (int)(i / Kpad);
    int k = (int)(i % Kpad);
    Wb[i] = (k < K) ? (bf16_t)W[(size_t)n * K + k] : (bf16_t)0.0f;
  }
}

// ---------------------------------------------------------------- generic GEMM
// C[M x N] = A[M x K]bf16 * W[N x K]bf16^T (+bias, optional relu), M = U_
// 256 threads / 8 waves; block tile 128x128; wave tile 32x64; double-buffered
// LDS fed by async global->LDS DMA; one barrier per 32-K step.

#define BM 128
#define BN 128
#define BK 32
#define LDA_ 40   // padded LDS stride (halves), 16B aligned

__global__ __launch_bounds__(256) void gemm_bf16_kernel(const bf16_t* __restrict__ A,
                                                        const bf16_t* __restrict__ W,
                                                        const float* __restrict__ bias,
                                                        float* __restrict__ C,
                                                        int N, int K, int relu) {
  __shared__ __attribute__((aligned(16))) bf16_t As[2][BM * LDA_];
  __shared__ __attribute__((aligned(16))) bf16_t Bs[2][BN * LDA_];

  const int bm   = blockIdx.x * BM;
  const int bn   = blockIdx.y * BN;
  const int tid  = threadIdx.x;
  const int wave = tid >> 5;
  const int lane = tid & 31;
  const int wm   = (wave & 3) * 32;   // 4 waves along M
  const int wn   = (wave >> 2) * 64;  // 2 waves along N

  auto issue_tile = [&](int k0, int buf) {
    // A tile 128x32 (512 x 16B) + B tile 128x32 (512 x 16B); 4 chunks/thread
#pragma unroll
    for (int it = 0; it < 2; ++it) {
      int id  = tid + it * 256;
      int row = id >> 2;
      int ch  = id & 3;
      async_load_b128(&As[buf][row * LDA_ + ch * 8],
                      A + (size_t)(bm + row) * K + k0 + ch * 8);
      async_load_b128(&Bs[buf][row * LDA_ + ch * 8],
                      W + (size_t)(bn + row) * K + k0 + ch * 8);
    }
  };

  v8f acc[2][4];
#pragma unroll
  for (int i = 0; i < 2; ++i)
#pragma unroll
    for (int j = 0; j < 4; ++j) acc[i][j] = v8f{0.f,0.f,0.f,0.f,0.f,0.f,0.f,0.f};

  issue_tile(0, 0);
  wait_async0();
  __syncthreads();

  int cur = 0;
  for (int k0 = 0; k0 < K; k0 += BK) {
    if (k0 + BK < K) issue_tile(k0 + BK, cur ^ 1);

    v16bf af0 = load_fragA(&As[cur][(wm +  0) * LDA_], LDA_);
    v16bf af1 = load_fragA(&As[cur][(wm + 16) * LDA_], LDA_);
    v16bf bfr[4];
#pragma unroll
    for (int j = 0; j < 4; ++j)
      bfr[j] = load_fragB(&Bs[cur][(wn + j * 16) * LDA_], LDA_);

#pragma unroll
    for (int j = 0; j < 4; ++j) {
      acc[0][j] = wmma_bf16(af0, bfr[j], acc[0][j]);
      acc[1][j] = wmma_bf16(af1, bfr[j], acc[1][j]);
    }

    wait_async0();     // next tile landed in LDS
    __syncthreads();   // publish to all waves / all done reading cur
    cur ^= 1;
  }

  // epilogue: C layout lane=N, vgpr r -> row r + 8*(lane>=16)
  const int cn = lane & 15;
  const int rb = (lane >> 4) * 8;
#pragma unroll
  for (int i = 0; i < 2; ++i) {
#pragma unroll
    for (int j = 0; j < 4; ++j) {
      int gcol = bn + wn + j * 16 + cn;
      float bv = bias ? bias[gcol] : 0.0f;
#pragma unroll
      for (int r = 0; r < 8; ++r) {
        int grow = bm + wm + i * 16 + rb + r;
        float v  = acc[i][j][r] + bv;
        if (relu) v = fmaxf(v, 0.0f);
        C[(size_t)grow * N + gcol] = v;
      }
    }
  }
}

// ---------------------------------------------------------------- elementwise kernels

__global__ __launch_bounds__(256) void gather_pad_kernel(const float* __restrict__ node_emb,
                                                         const int* __restrict__ tile_ids,
                                                         const float* __restrict__ ent,
                                                         bf16_t* __restrict__ Ain) {
  int u = blockIdx.x;
  int t = threadIdx.x;
  int id = tile_ids[u];
  const float* src = node_emb + (size_t)id * D_;
  bf16_t* dst = Ain + (size_t)u * KPADI;
  for (int c = t; c < D_; c += 256) dst[c] = (bf16_t)src[c];
  if (t == 0) dst[D_] = (bf16_t)ent[u];
  if (t >= 1 && t < 32) dst[D_ + t] = (bf16_t)0.0f;  // pad 1025..1055
}

__global__ __launch_bounds__(256) void copy_f32_bf16_kernel(const float* __restrict__ in,
                                                            float* __restrict__ of,
                                                            bf16_t* __restrict__ ob,
                                                            size_t n) {
  size_t i = (size_t)blockIdx.x * 256 + threadIdx.x;
  if (i < n) { float v = in[i]; of[i] = v; ob[i] = (bf16_t)v; }
}

__global__ __launch_bounds__(256) void f32_to_bf16_kernel(const float* __restrict__ in,
                                                          bf16_t* __restrict__ out,
                                                          size_t n) {
  size_t i = (size_t)blockIdx.x * 256 + threadIdx.x;
  if (i < n) out[i] = (bf16_t)in[i];
}

__global__ __launch_bounds__(256) void qkv_conv_kernel(const float* __restrict__ yq,
                                                       const float* __restrict__ yk,
                                                       const float* __restrict__ yv,
                                                       bf16_t* __restrict__ qb,
                                                       bf16_t* __restrict__ kb,
                                                       bf16_t* __restrict__ vb,
                                                       size_t n) {
  size_t i = (size_t)blockIdx.x * 256 + threadIdx.x;
  if (i < n) {
    qb[i] = (bf16_t)(yq[i] * 0.125f);  // fold 1/sqrt(64)
    kb[i] = (bf16_t)yk[i];
    vb[i] = (bf16_t)yv[i];
  }
}

// RoPE: xr = [rope(x[:512], row), rope(x[512:], col)] -> bf16
__global__ __launch_bounds__(256) void rope_kernel(const float* __restrict__ x,
                                                   const int* __restrict__ tile_ids,
                                                   const int* __restrict__ Ny_p,
                                                   bf16_t* __restrict__ xr) {
  int u = blockIdx.x;
  int Ny = Ny_p[0];
  int idv = tile_ids[u];
  float rowp = (float)(idv / Ny);
  float colp = (float)(idv - (idv / Ny) * Ny);
  const float* xrow = x + (size_t)u * D_;
  bf16_t* orow = xr + (size_t)u * D_;
  for (int c = threadIdx.x; c < D_; c += 256) {
    int bh = c >> 9;          // which 512-block
    int ci = c & 511;
    int hi = ci & 255;        // theta index
    float pos   = bh ? colp : rowp;
    float theta = __powf(10000.0f, -(float)hi * (1.0f / 256.0f));
    float sn, cs;
    __sincosf(pos * theta, &sn, &cs);
    const float* xb = xrow + (bh << 9);
    float x1 = xb[hi], x2 = xb[hi + 256];
    float v = (ci < 256) ? (x1 * cs - x2 * sn) : (x1 * sn + x2 * cs);
    orow[c] = (bf16_t)v;
  }
}

// x = LN(x + y)*g + b ; write f32 and bf16 copies. One wave per row.
__global__ __launch_bounds__(256) void add_ln_kernel(const float* __restrict__ x,
                                                     const float* __restrict__ y,
                                                     const float* __restrict__ g,
                                                     const float* __restrict__ b,
                                                     float* __restrict__ xout,
                                                     bf16_t* __restrict__ bout) {
  int wave = threadIdx.x >> 5, lane = threadIdx.x & 31;
  int row  = blockIdx.x * 8 + wave;
  const float* xr = x + (size_t)row * D_;
  const float* yr = y + (size_t)row * D_;
  float vals[32];
  float s = 0.0f;
#pragma unroll
  for (int i = 0; i < 32; ++i) {
    float v = xr[lane + 32 * i] + yr[lane + 32 * i];
    vals[i] = v; s += v;
  }
#pragma unroll
  for (int off = 16; off >= 1; off >>= 1) s += __shfl_xor(s, off, 32);
  float mean = s * (1.0f / D_);
  float var = 0.0f;
#pragma unroll
  for (int i = 0; i < 32; ++i) { float d = vals[i] - mean; var += d * d; }
#pragma unroll
  for (int off = 16; off >= 1; off >>= 1) var += __shfl_xor(var, off, 32);
  float inv = rsqrtf(var * (1.0f / D_) + 1e-5f);
#pragma unroll
  for (int i = 0; i < 32; ++i) {
    int c = lane + 32 * i;
    float o = (vals[i] - mean) * inv * g[c] + b[c];
    xout[(size_t)row * D_ + c] = o;
    bout[(size_t)row * D_ + c] = (bf16_t)o;
  }
}

// ---------------------------------------------------------------- flash attention
// grid (U/64, H), 128 threads (4 waves). Each wave owns 16 full query rows.
// Double-buffered K/V tiles; K tile via async global->LDS DMA.

#define KB   64   // key block
#define LDT  72   // padded LDS stride (halves)

__global__ __launch_bounds__(128) void flash_attn_kernel(const bf16_t* __restrict__ Q,
                                                         const bf16_t* __restrict__ Km,
                                                         const bf16_t* __restrict__ Vm,
                                                         float* __restrict__ O) {
  __shared__ __attribute__((aligned(16))) bf16_t Kt[2][KB * LDT];      // [key][feat]
  __shared__ __attribute__((aligned(16))) bf16_t Vt[2][HD_ * LDT];     // [feat][key]
  __shared__ __attribute__((aligned(16))) bf16_t Ps[4 * 16 * LDT];     // per-wave [row][key]

  const int head  = blockIdx.y;
  const int qbase = blockIdx.x * 64;
  const int tid   = threadIdx.x;
  const int wave  = tid >> 5;
  const int lane  = tid & 31;
  const int cn    = lane & 15;
  const int rb    = (lane >> 4) * 8;
  const int fcol  = head * HD_;

  auto load_tiles = [&](int kb, int buf) {
    // K tile: 64x64 bf16 = 512 x 16B chunks, 4 per thread (async DMA)
#pragma unroll
    for (int it = 0; it < 4; ++it) {
      int id  = tid + it * 128;
      int row = id >> 3;
      int ch  = id & 7;
      async_load_b128(&Kt[buf][row * LDT + ch * 8],
                      Km + (size_t)(kb + row) * D_ + fcol + ch * 8);
    }
    // V tile transposed [feat][key]; pack key pairs -> b32 LDS stores
#pragma unroll
    for (int it = 0; it < 2; ++it) {
      int task = tid + it * 128;        // 256 tasks: 32 key-pairs x 8 feat-groups
      int kp   = task & 31;
      int fg   = task >> 5;
      int key  = kp * 2;
      const bf16_t* v0 = Vm + (size_t)(kb + key) * D_ + fcol + fg * 8;
      v8bf a = *reinterpret_cast<const v8bf*>(v0);
      v8bf b = *reinterpret_cast<const v8bf*>(v0 + D_);
#pragma unroll
      for (int j = 0; j < 8; ++j) {
        union { bf16_t h[2]; unsigned int u; } p;
        p.h[0] = a[j]; p.h[1] = b[j];
        *reinterpret_cast<unsigned int*>(&Vt[buf][(fg * 8 + j) * LDT + key]) = p.u;
      }
    }
  };

  // Q fragments for this wave's 16 rows (2 K-steps of 32)
  const bf16_t* qb = Q + (size_t)(qbase + wave * 16) * D_ + fcol;
  v16bf aq0 = load_fragA(qb, D_);
  v16bf aq1 = load_fragA(qb + 32, D_);

  v8f o[4];
#pragma unroll
  for (int i = 0; i < 4; ++i) o[i] = v8f{0.f,0.f,0.f,0.f,0.f,0.f,0.f,0.f};
  float mstat[8], lstat[8];
#pragma unroll
  for (int r = 0; r < 8; ++r) { mstat[r] = -3.0e38f; lstat[r] = 0.0f; }

  load_tiles(0, 0);
  wait_async0();
  __syncthreads();

  int cur = 0;
  for (int kb = 0; kb < U_; kb += KB) {
    if (kb + KB < U_) load_tiles(kb + KB, cur ^ 1);

    // S = Q * K^T for this wave's 16 rows x 64 keys
    v8f s[4];
#pragma unroll
    for (int nt = 0; nt < 4; ++nt) {
      v8f acc = v8f{0.f,0.f,0.f,0.f,0.f,0.f,0.f,0.f};
      v16bf b0 = load_fragB(&Kt[cur][nt * 16 * LDT], LDT);
      v16bf b1 = load_fragB(&Kt[cur][nt * 16 * LDT + 32], LDT);
      acc = wmma_bf16(aq0, b0, acc);
      acc = wmma_bf16(aq1, b1, acc);
      s[nt] = acc;
    }

    // online softmax per row (rows = rb + r for this lane half)
#pragma unroll
    for (int r = 0; r < 8; ++r) {
      float rmax = fmaxf(fmaxf(s[0][r], s[1][r]), fmaxf(s[2][r], s[3][r]));
#pragma unroll
      for (int off = 8; off >= 1; off >>= 1) rmax = fmaxf(rmax, __shfl_xor(rmax, off, 16));
      float newm  = fmaxf(mstat[r], rmax);
      float scale = __expf(mstat[r] - newm);
      mstat[r] = newm;
      float rsum = 0.0f;
#pragma unroll
      for (int nt = 0; nt < 4; ++nt) {
        float p = __expf(s[nt][r] - newm);
        s[nt][r] = p;
        rsum += p;
      }
#pragma unroll
      for (int off = 8; off >= 1; off >>= 1) rsum += __shfl_xor(rsum, off, 16);
      lstat[r] = lstat[r] * scale + rsum;
#pragma unroll
      for (int nt = 0; nt < 4; ++nt) o[nt][r] *= scale;
    }

    // write P (bf16) to this wave's LDS slice: [row 0..15][key 0..63]
    bf16_t* ps = Ps + wave * 16 * LDT;
#pragma unroll
    for (int nt = 0; nt < 4; ++nt) {
#pragma unroll
      for (int r = 0; r < 8; ++r) {
        ps[(rb + r) * LDT + nt * 16 + cn] = (bf16_t)s[nt][r];
      }
    }

    // O += P * V   (A = P rows, B = Vt rows = V columns)
#pragma unroll
    for (int kk = 0; kk < 2; ++kk) {
      v16bf ap = load_fragA(ps + kk * 32, LDT);
#pragma unroll
      for (int nt = 0; nt < 4; ++nt) {
        v16bf bv = load_fragB(&Vt[cur][nt * 16 * LDT + kk * 32], LDT);
        o[nt] = wmma_bf16(ap, bv, o[nt]);
      }
    }

    wait_async0();
    __syncthreads();
    cur ^= 1;
  }

  // normalize and store O (f32, [unit][D], this head's 64 columns)
#pragma unroll
  for (int nt = 0; nt < 4; ++nt) {
#pragma unroll
    for (int r = 0; r < 8; ++r) {
      int grow = qbase + wave * 16 + rb + r;
      int gcol = fcol + nt * 16 + cn;
      O[(size_t)grow * D_ + gcol] = o[nt][r] / lstat[r];
    }
  }
}

// ---------------------------------------------------------------- head logits + softmax

__global__ __launch_bounds__(256) void logits_kernel(const float* __restrict__ h1,
                                                     const float* __restrict__ w2,
                                                     const float* __restrict__ b2,
                                                     float* __restrict__ out) {
  int wave = threadIdx.x >> 5, lane = threadIdx.x & 31;
  int row  = blockIdx.x * 8 + wave;
  const float* hr = h1 + (size_t)row * HID_;
  float s = 0.0f;
#pragma unroll
  for (int i = lane; i < HID_; i += 32) s += hr[i] * w2[i];
#pragma unroll
  for (int off = 16; off >= 1; off >>= 1) s += __shfl_xor(s, off, 32);
  if (lane == 0) out[row] = s + b2[0];
}

__global__ __launch_bounds__(1024) void softmax_entropy_kernel(const float* __restrict__ logits,
                                                               float* __restrict__ dout) {
  __shared__ float red[32];
  int t = threadIdx.x, wave = t >> 5, lane = t & 31;
  float lv[4];
  float m = -3.0e38f;
#pragma unroll
  for (int i = 0; i < 4; ++i) { lv[i] = logits[t + 1024 * i]; m = fmaxf(m, lv[i]); }
#pragma unroll
  for (int off = 16; off >= 1; off >>= 1) m = fmaxf(m, __shfl_xor(m, off, 32));
  if (lane == 0) red[wave] = m;
  __syncthreads();
  if (wave == 0) {
    float v = red[lane];
#pragma unroll
    for (int off = 16; off >= 1; off >>= 1) v = fmaxf(v, __shfl_xor(v, off, 32));
    if (lane == 0) red[0] = v;
  }
  __syncthreads();
  m = red[0];
  __syncthreads();

  float z = 0.0f;
#pragma unroll
  for (int i = 0; i < 4; ++i) z += __expf(lv[i] - m);
#pragma unroll
  for (int off = 16; off >= 1; off >>= 1) z += __shfl_xor(z, off, 32);
  if (lane == 0) red[wave] = z;
  __syncthreads();
  if (wave == 0) {
    float v = red[lane];
#pragma unroll
    for (int off = 16; off >= 1; off >>= 1) v += __shfl_xor(v, off, 32);
    if (lane == 0) red[0] = v;
  }
  __syncthreads();
  z = red[0];
  __syncthreads();

  float invz = 1.0f / z;
  float ent = 0.0f;
#pragma unroll
  for (int i = 0; i < 4; ++i) {
    float p = __expf(lv[i] - m) * invz;
    dout[t + 1024 * i] = p;                 // probs
    dout[4097 + t + 1024 * i] = lv[i];      // logits
    ent += -p * __logf(p + 1e-12f);
  }
#pragma unroll
  for (int off = 16; off >= 1; off >>= 1) ent += __shfl_xor(ent, off, 32);
  if (lane == 0) red[wave] = ent;
  __syncthreads();
  if (wave == 0) {
    float v = red[lane];
#pragma unroll
    for (int off = 16; off >= 1; off >>= 1) v += __shfl_xor(v, off, 32);
    if (lane == 0) dout[4096] = v;          // entropy
  }
}

// ---------------------------------------------------------------- host orchestration

extern "C" void kernel_launch(void* const* d_in, const int* in_sizes, int n_in,
                              void* d_out, int out_size, void* d_ws, size_t ws_size,
                              hipStream_t stream) {
  const float* node_emb  = (const float*)d_in[0];
  const int*   tile_ids  = (const int*)d_in[1];
  const float* entropies = (const float*)d_in[2];
  const int*   Ny        = (const int*)d_in[3];
  const float* in_w      = (const float*)d_in[4];
  const float* in_b      = (const float*)d_in[5];
  const float* attn_in_w = (const float*)d_in[6];
  const float* attn_in_b = (const float*)d_in[7];
  const float* attn_out_w= (const float*)d_in[8];
  const float* attn_out_b= (const float*)d_in[9];
  const float* ff_w1     = (const float*)d_in[10];
  const float* ff_b1     = (const float*)d_in[11];
  const float* ff_w2     = (const float*)d_in[12];
  const float* ff_b2     = (const float*)d_in[13];
  const float* ln1_g     = (const float*)d_in[14];
  const float* ln1_b     = (const float*)d_in[15];
  const float* ln2_g     = (const float*)d_in[16];
  const float* ln2_b     = (const float*)d_in[17];
  const float* sh_w0     = (const float*)d_in[18];
  const float* sh_b0     = (const float*)d_in[19];
  const float* sh_w1     = (const float*)d_in[20];
  const float* sh_b1     = (const float*)d_in[21];
  const float* sh_w2     = (const float*)d_in[22];
  const float* sh_b2     = (const float*)d_in[23];

  char* w = (char*)d_ws;
  float*  xf   = (float*) (w + 0);                    // U x D f32 (residual)
  float*  yf   = (float*) (w + (size_t)16777216);     // U x 3D f32 scratch
  bf16_t* abf  = (bf16_t*)(w + (size_t)67108864);     // U x 2048 bf16 A-staging
  bf16_t* xbf  = (bf16_t*)(w + (size_t)83886080);     // U x D bf16 (x copy)
  bf16_t* qbf  = (bf16_t*)(w + (size_t)92274688);
  bf16_t* kbf  = (bf16_t*)(w + (size_t)100663296);
  bf16_t* vbf  = (bf16_t*)(w + (size_t)109051904);
  bf16_t* wbf  = (bf16_t*)(w + (size_t)117440512);    // 3D x D bf16 weight staging
  float*  lgts = (float*) (w + (size_t)123731968);    // U logits

  const size_t UD = (size_t)U_ * D_;

  auto conv_w = [&](const float* Wf, int N, int K, int Kpad) {
    size_t total = (size_t)N * Kpad;
    int blocks = (int)((total + 255) / 256);
    convw_kernel<<<blocks, 256, 0, stream>>>(Wf, wbf, N, K, Kpad);
  };
  auto gemm = [&](const bf16_t* A, const bf16_t* W, const float* bias,
                  float* C, int N, int Kpad, int relu) {
    dim3 g(U_ / BM, N / BN);
    gemm_bf16_kernel<<<g, 256, 0, stream>>>(A, W, bias, C, N, Kpad, relu);
  };

  // ---- input projection: x = concat(emb[ids], ent) @ in_w.T + in_b
  gather_pad_kernel<<<U_, 256, 0, stream>>>(node_emb, tile_ids, entropies, abf);
  conv_w(in_w, D_, D_ + 1, KPADI);
  gemm(abf, wbf, in_b, yf, D_, KPADI, 0);
  copy_f32_bf16_kernel<<<(int)(UD / 256), 256, 0, stream>>>(yf, xf, xbf, UD);

  // ---- transformer layers
  for (int l = 0; l < L_; ++l) {
    // RoPE (xr -> abf bf16)
    rope_kernel<<<U_, 256, 0, stream>>>(xf, tile_ids, Ny, abf);

    // QKV projections
    conv_w(attn_in_w + (size_t)l * 3 * D_ * D_, 3 * D_, D_, D_);
    const float* bqkv = attn_in_b + (size_t)l * 3 * D_;
    gemm(abf, wbf,                        bqkv,          yf,           D_, D_, 0); // q from xr
    gemm(abf, wbf + (size_t)D_ * D_,      bqkv + D_,     yf + UD,      D_, D_, 0); // k from xr
    gemm(xbf, wbf + (size_t)2 * D_ * D_,  bqkv + 2 * D_, yf + 2 * UD,  D_, D_, 0); // v from x

    qkv_conv_kernel<<<(int)(UD / 256), 256, 0, stream>>>(yf, yf + UD, yf + 2 * UD,
                                                         qbf, kbf, vbf, UD);

    // attention -> O f32 in yf
    flash_attn_kernel<<<dim3(U_ / 64, H_), 128, 0, stream>>>(qbf, kbf, vbf, yf);

    // out projection
    f32_to_bf16_kernel<<<(int)(UD / 256), 256, 0, stream>>>(yf, abf, UD);
    conv_w(attn_out_w + (size_t)l * D_ * D_, D_, D_, D_);
    gemm(abf, wbf, attn_out_b + (size_t)l * D_, yf + UD, D_, D_, 0);

    // x = LN1(x + o)  (bf16 copy -> abf for FF input)
    add_ln_kernel<<<U_ / 8, 256, 0, stream>>>(xf, yf + UD,
                                              ln1_g + (size_t)l * D_, ln1_b + (size_t)l * D_,
                                              xf, abf);

    // FF: h = relu(x @ w1.T + b1); f = h @ w2.T + b2
    conv_w(ff_w1 + (size_t)l * 2 * D_ * D_, 2 * D_, D_, D_);
    gemm(abf, wbf, ff_b1 + (size_t)l * 2 * D_, yf, 2 * D_, D_, 1);
    f32_to_bf16_kernel<<<(int)(2 * UD / 256), 256, 0, stream>>>(yf, abf, 2 * UD);
    conv_w(ff_w2 + (size_t)l * D_ * 2 * D_, D_, 2 * D_, 2 * D_);
    gemm(abf, wbf, ff_b2 + (size_t)l * D_, yf + 2 * UD, D_, 2 * D_, 0);

    // x = LN2(x + f)  (bf16 copy -> xbf for next layer V-proj / head)
    add_ln_kernel<<<U_ / 8, 256, 0, stream>>>(xf, yf + 2 * UD,
                                              ln2_g + (size_t)l * D_, ln2_b + (size_t)l * D_,
                                              xf, xbf);
  }

  // ---- MLP head
  const size_t UH = (size_t)U_ * HID_;
  conv_w(sh_w0, HID_, D_, D_);
  gemm(xbf, wbf, sh_b0, yf, HID_, D_, 1);                 // h0 = relu(...)
  f32_to_bf16_kernel<<<(int)(UH / 256), 256, 0, stream>>>(yf, abf, UH);
  conv_w(sh_w1, HID_, HID_, HID_);
  gemm(abf, wbf, sh_b1, yf + UH, HID_, HID_, 1);          // h1 = relu(...)
  logits_kernel<<<U_ / 8, 256, 0, stream>>>(yf + UH, sh_w2, sh_b2, lgts);

  softmax_entropy_kernel<<<1, 1024, 0, stream>>>(lgts, (float*)d_out);
}
